// TransformerBlock_57956288692459
// MI455X (gfx1250) — matrix operand order repaired and checked by reading, not compiled, not run on hardware
//
#include <hip/hip_runtime.h>

typedef __bf16 bf16;
typedef __attribute__((ext_vector_type(8)))  bf16  v8bf;
typedef __attribute__((ext_vector_type(16))) bf16  v16bf;
typedef __attribute__((ext_vector_type(8)))  float v8f;

#define B_  2
#define S_  2048
#define D_  1024
#define F_  4096
#define H_  16
#define DH_ 64
#define M_  (B_ * S_)

__device__ __forceinline__ bf16 f2bf(float f) {
  unsigned u = __builtin_bit_cast(unsigned, f);
  unsigned r = (u + 0x7FFFu + ((u >> 16) & 1u)) >> 16;   // round-to-nearest-even
  unsigned short s = (unsigned short)r;
  return __builtin_bit_cast(bf16, s);
}

__device__ __forceinline__ v16bf cat8(v8bf lo, v8bf hi) {
  return __builtin_shufflevector(lo, hi, 0,1,2,3,4,5,6,7,8,9,10,11,12,13,14,15);
}

// A fragment (16x32 bf16) from row-major [16][ld].
__device__ __forceinline__ v16bf load_a_frag(const bf16* base, int ld, int lane) {
  int r = lane & 15, hi = lane >> 4;
  const bf16* p = base + (size_t)r * ld;
  return *(const v16bf*)(p + hi * 16);
}

// B fragment (32x16 bf16) from K-contiguous-per-column storage BT[n][ld].
__device__ __forceinline__ v16bf load_b_frag(const bf16* baseT, int ld, int lane) {
  int n = lane & 15, hi = lane >> 4;
  return *(const v16bf*)(baseT + (size_t)n * ld + hi * 16);
}

#define WMMA(a, b, c) \
  __builtin_amdgcn_wmma_f32_16x16x32_bf16(false, (a), false, (b), (short)0, (c), false, false)

// Async global->LDS copy of 16 bytes (GVS mode). Tracked by ASYNCcnt.
__device__ __forceinline__ void async_b128(void* lds_ptr, int gbyte_off, const void* sbase) {
  unsigned lds_off = (unsigned)(unsigned long long)lds_ptr;  // low 32b of LDS aperture addr
  asm volatile("global_load_async_to_lds_b128 %0, %1, %2"
               :: "v"(lds_off), "v"(gbyte_off), "s"(sbase) : "memory");
}
__device__ __forceinline__ void wait_async0() {
  asm volatile("s_wait_asynccnt 0x0" ::: "memory");
}

// ---- 16-lane (DPP-row) VALU reductions: xor1, xor2, half-mirror, mirror ----
template <int CTRL>
__device__ __forceinline__ float dpp_bcast(float x) {
  int t = __builtin_amdgcn_update_dpp(0, __builtin_bit_cast(int, x), CTRL, 0xF, 0xF, true);
  return __builtin_bit_cast(float, t);
}
__device__ __forceinline__ float red16_max(float x) {
  x = fmaxf(x, dpp_bcast<0xB1>(x));    // quad_perm [1,0,3,2]
  x = fmaxf(x, dpp_bcast<0x4E>(x));    // quad_perm [2,3,0,1]
  x = fmaxf(x, dpp_bcast<0x141>(x));   // row_half_mirror
  x = fmaxf(x, dpp_bcast<0x140>(x));   // row_mirror
  return x;
}
__device__ __forceinline__ float red16_sum(float x) {
  x += dpp_bcast<0xB1>(x);
  x += dpp_bcast<0x4E>(x);
  x += dpp_bcast<0x141>(x);
  x += dpp_bcast<0x140>(x);
  return x;
}

// ---------------------------------------------------------------------------
// weight prep: f32 [K][N] -> bf16 transposed [N][K] (LDS-tiled, coalesced)
// ---------------------------------------------------------------------------
__global__ __launch_bounds__(256) void transpose_cast_bf16(const float* __restrict__ in,
                                                           bf16* __restrict__ out,
                                                           int K, int N) {
  __shared__ float t[64][65];
  int kt = blockIdx.x * 64, nt = blockIdx.y * 64;
  int row = threadIdx.x >> 2, c = (threadIdx.x & 3) << 4;
  const float* src = in + (size_t)(kt + row) * N + nt + c;
#pragma unroll
  for (int e = 0; e < 16; e += 4) {
    float4 v = *(const float4*)(src + e);
    t[row][c + e + 0] = v.x; t[row][c + e + 1] = v.y;
    t[row][c + e + 2] = v.z; t[row][c + e + 3] = v.w;
  }
  __syncthreads();
  bf16* dst = out + (size_t)(nt + row) * K + kt + c;
#pragma unroll
  for (int e = 0; e < 16; ++e) dst[e] = f2bf(t[c + e][row]);
}

// ---------------------------------------------------------------------------
// LayerNorm: f32 [rows][1024] -> bf16, one block per row
// ---------------------------------------------------------------------------
__global__ __launch_bounds__(256) void layernorm_bf16_k(const float* __restrict__ x,
                                                        const float* __restrict__ sc,
                                                        const float* __restrict__ bi,
                                                        bf16* __restrict__ out) {
  int row = blockIdx.x, tid = threadIdx.x;
  __shared__ float rs[256], rq[256];
  const float* xr = x + (size_t)row * D_;
  float vals[4];
  float s = 0.f, s2 = 0.f;
#pragma unroll
  for (int i = 0; i < 4; ++i) {
    float v = xr[tid + i * 256];
    vals[i] = v; s += v; s2 += v * v;
  }
  rs[tid] = s; rq[tid] = s2;
  __syncthreads();
  for (int st = 128; st > 0; st >>= 1) {
    if (tid < st) { rs[tid] += rs[tid + st]; rq[tid] += rq[tid + st]; }
    __syncthreads();
  }
  float mu = rs[0] * (1.0f / D_);
  float var = rq[0] * (1.0f / D_) - mu * mu;
  float rstd = rsqrtf(var + 1e-6f);
#pragma unroll
  for (int i = 0; i < 4; ++i) {
    int c = tid + i * 256;
    out[(size_t)row * D_ + c] = f2bf((vals[i] - mu) * rstd * sc[c] + bi[c]);
  }
}

// ---------------------------------------------------------------------------
// bf16 WMMA GEMM: C = A[MxK] @ B (given as BT[N][K]) + bias (+res/relu/scale)
// 128x128 macro-tile, BLOCK_K=64 (2 WMMA k-steps), 8 waves, each 32x64.
// Double-buffered LDS tiles filled with async global->LDS loads.
// ---------------------------------------------------------------------------
template <bool RELU, bool RES, bool OUTF, bool OUTB, bool TRANSB_OUT, bool SCALE>
__global__ __launch_bounds__(256) void gemm_bf16_wmma(
    const bf16* __restrict__ A, const bf16* __restrict__ BT,
    const float* __restrict__ bias, const float* __restrict__ residual,
    float* __restrict__ outF, bf16* __restrict__ outB,
    int M, int N, int K) {
  __shared__ __attribute__((aligned(32))) bf16 ldsA[2][128 * 64];
  __shared__ __attribute__((aligned(32))) bf16 ldsB[2][128 * 64];  // [n][k] tiles

  const int tid = threadIdx.x;
  const int lane = tid & 31, wave = tid >> 5;
  const int hi = lane >> 4, l15 = lane & 15;
  const int wm = wave >> 1, wn = wave & 1;
  const int rm = wm * 32, cn = wn * 64;
  const int m0 = blockIdx.y * 128, n0 = blockIdx.x * 128;

  const int srow = tid >> 1, sc32 = (tid & 1) << 5;  // staging: 128 rows x 64 cols

  auto stage = [&](int buf, int k0) {
    bf16* la = &ldsA[buf][srow * 64 + sc32];
    int ga = ((m0 + srow) * K + k0 + sc32) * 2;
    bf16* lb = &ldsB[buf][srow * 64 + sc32];
    int gb = ((n0 + srow) * K + k0 + sc32) * 2;
#pragma unroll
    for (int c = 0; c < 32; c += 8) {
      async_b128(la + c, ga + c * 2, A);
      async_b128(lb + c, gb + c * 2, BT);
    }
  };

  v8f vz = {};
  v8f acc[2][4];
#pragma unroll
  for (int i = 0; i < 2; ++i)
#pragma unroll
    for (int j = 0; j < 4; ++j) acc[i][j] = vz;

  const int nk = K >> 6;
  stage(0, 0);
  for (int it = 0; it < nk; ++it) {
    const int cur = it & 1;
    wait_async0();
    __syncthreads();                       // tile [cur] complete, visible to all
    if (it + 1 < nk) stage(cur ^ 1, (it + 1) * 64);

#pragma unroll
    for (int kk = 0; kk < 2; ++kk) {
      v16bf af[2], bfr[4];
#pragma unroll
      for (int i = 0; i < 2; ++i)
        af[i] = load_a_frag(&ldsA[cur][(rm + i * 16) * 64 + kk * 32], 64, lane);
#pragma unroll
      for (int j = 0; j < 4; ++j)
        bfr[j] = load_b_frag(&ldsB[cur][(cn + j * 16) * 64 + kk * 32], 64, lane);
#pragma unroll
      for (int i = 0; i < 2; ++i)
#pragma unroll
        for (int j = 0; j < 4; ++j)
          acc[i][j] = WMMA(af[i], bfr[j], acc[i][j]);
    }
    __syncthreads();                       // all reads of [cur] done before re-fill
  }

  // epilogue (straight-line, compile-time mode)
#pragma unroll
  for (int i = 0; i < 2; ++i)
#pragma unroll
    for (int j = 0; j < 4; ++j) {
      int rbase = m0 + rm + i * 16 + hi * 8;
      int col = n0 + cn + j * 16 + l15;
      float bia = bias[col];
#pragma unroll
      for (int v = 0; v < 8; ++v) {
        int row = rbase + v;
        float val = acc[i][j][v] + bia;
        if constexpr (RES)   val += residual[(size_t)row * N + col];
        if constexpr (RELU)  val = fmaxf(val, 0.f);
        if constexpr (SCALE) val *= 0.125f;             // fold 1/sqrt(Dh) into Q
        if constexpr (OUTF)  outF[(size_t)row * N + col] = val;
        if constexpr (OUTB) {
          if constexpr (TRANSB_OUT) outB[(size_t)col * M + row] = f2bf(val);
          else                      outB[(size_t)row * N + col] = f2bf(val);
        }
      }
    }
}

// ---------------------------------------------------------------------------
// Flash attention, one WG per (128 q-rows, head, batch). 8 waves x 16 rows.
// Q/K: bf16 [B*S][1024] (head offset h*64); Q is pre-scaled by 1/sqrt(Dh).
// V: bf16 transposed [1024][B*S]. K/V tiles double-buffered via async loads.
// ---------------------------------------------------------------------------
__global__ __launch_bounds__(256) void flash_attn_wmma(
    const bf16* __restrict__ Q, const bf16* __restrict__ Km,
    const bf16* __restrict__ VT, bf16* __restrict__ Out) {
  __shared__ __attribute__((aligned(32))) bf16 Kt[2][64 * 64];   // [key][d]
  __shared__ __attribute__((aligned(32))) bf16 Vt[2][64 * 64];   // [d][key]
  __shared__ __attribute__((aligned(32))) bf16 Pb[8 * 16 * 64];  // per-wave P

  const int tid = threadIdx.x;
  const int lane = tid & 31, wave = tid >> 5;
  const int hi = lane >> 4, l15 = lane & 15;
  const int h = blockIdx.y, b = blockIdx.z;
  const int q0 = blockIdx.x * 128 + wave * 16;
  const size_t ld = D_;
  const int hoff = h * DH_;
  const size_t bs = (size_t)b * S_;

  // Q fragments, contraction Dh=64 -> 2 k-steps, held in registers
  v16bf qf[2];
#pragma unroll
  for (int kk = 0; kk < 2; ++kk)
    qf[kk] = load_a_frag(Q + (bs + q0) * ld + hoff + kk * 32, (int)ld, lane);

  v8f vz = {};
  v8f o[4];
#pragma unroll
  for (int j = 0; j < 4; ++j) o[j] = vz;
  float m8[8], l8[8];
#pragma unroll
  for (int v = 0; v < 8; ++v) { m8[v] = -__builtin_inff(); l8[v] = 0.f; }

  const int srow = tid >> 2;        // 0..63 (staging)
  const int scol = (tid & 3) << 4;  // 0,16,32,48
  bf16* Pw = Pb + wave * (16 * 64);

  auto stage = [&](int buf, int kb) {
    bf16* lk = &Kt[buf][srow * 64 + scol];
    int gk = (int)(((bs + kb + srow) * ld + hoff + scol) * 2);
    async_b128(lk, gk, Km);
    async_b128(lk + 8, gk + 16, Km);
    bf16* lv = &Vt[buf][srow * 64 + scol];
    int gv = (int)((((size_t)(hoff + srow)) * M_ + bs + kb + scol) * 2);
    async_b128(lv, gv, VT);
    async_b128(lv + 8, gv + 16, VT);
  };

  const int nkb = S_ / 64;
  stage(0, 0);
  for (int it = 0; it < nkb; ++it) {
    const int cur = it & 1;
    wait_async0();
    __syncthreads();
    if (it + 1 < nkb) stage(cur ^ 1, (it + 1) * 64);

    // scores = Q @ K^T (4 tiles of 16 keys x 2 k-steps); Q pre-scaled
    v8f s[4];
#pragma unroll
    for (int j = 0; j < 4; ++j) {
      s[j] = vz;
#pragma unroll
      for (int kk = 0; kk < 2; ++kk)
        s[j] = WMMA(qf[kk], load_b_frag(&Kt[cur][(j * 16) * 64 + kk * 32], 64, lane), s[j]);
    }

    // online softmax: per-row max/sum via DPP 16-lane butterflies
#pragma unroll
    for (int v = 0; v < 8; ++v) {
      float s0 = s[0][v], s1 = s[1][v], s2 = s[2][v], s3 = s[3][v];
      float mx = red16_max(fmaxf(fmaxf(s0, s1), fmaxf(s2, s3)));
      float mn = fmaxf(m8[v], mx);
      float alpha = __expf(m8[v] - mn);
      float p0 = __expf(s0 - mn), p1 = __expf(s1 - mn);
      float p2 = __expf(s2 - mn), p3 = __expf(s3 - mn);
      float rsum = red16_sum(p0 + p1 + p2 + p3);
      l8[v] = l8[v] * alpha + rsum;
      m8[v] = mn;
      s[0][v] = p0; s[1][v] = p1; s[2][v] = p2; s[3][v] = p3;
      o[0][v] *= alpha; o[1][v] *= alpha; o[2][v] *= alpha; o[3][v] *= alpha;
    }

    // reshape P: C-layout -> row-major bf16 in per-wave LDS (A-frag source)
#pragma unroll
    for (int v = 0; v < 8; ++v)
#pragma unroll
      for (int j = 0; j < 4; ++j)
        Pw[(v + hi * 8) * 64 + j * 16 + l15] = f2bf(s[j][v]);
    asm volatile("s_wait_dscnt 0x0" ::: "memory");  // same-wave LDS RAW

    // O += P @ V
#pragma unroll
    for (int kk = 0; kk < 2; ++kk) {
      v16bf pf = load_a_frag(Pw + kk * 32, 64, lane);
#pragma unroll
      for (int j = 0; j < 4; ++j)
        o[j] = WMMA(pf, load_b_frag(&Vt[cur][(j * 16) * 64 + kk * 32], 64, lane), o[j]);
    }
    __syncthreads();
  }

  // normalize and write bf16 output at [b*S+q][h*64+d]
#pragma unroll
  for (int j = 0; j < 4; ++j) {
    int col = hoff + j * 16 + l15;
#pragma unroll
    for (int v = 0; v < 8; ++v) {
      size_t row = bs + q0 + v + hi * 8;
      Out[row * ld + col] = f2bf(o[j][v] / l8[v]);
    }
  }
}

// ---------------------------------------------------------------------------
// host launcher
// ---------------------------------------------------------------------------
extern "C" void kernel_launch(void* const* d_in, const int* in_sizes, int n_in,
                              void* d_out, int out_size, void* d_ws, size_t ws_size,
                              hipStream_t stream) {
  (void)in_sizes; (void)n_in; (void)out_size; (void)ws_size;
  const float* x    = (const float*)d_in[0];
  const float* Wq   = (const float*)d_in[1];  const float* bq = (const float*)d_in[2];
  const float* Wk   = (const float*)d_in[3];  const float* bk = (const float*)d_in[4];
  const float* Wv   = (const float*)d_in[5];  const float* bv = (const float*)d_in[6];
  const float* Wo   = (const float*)d_in[7];  const float* bo = (const float*)d_in[8];
  const float* W1   = (const float*)d_in[9];  const float* b1 = (const float*)d_in[10];
  const float* W2   = (const float*)d_in[11]; const float* b2 = (const float*)d_in[12];
  const float* ln1s = (const float*)d_in[13]; const float* ln1b = (const float*)d_in[14];
  const float* ln2s = (const float*)d_in[15]; const float* ln2b = (const float*)d_in[16];
  float* out = (float*)d_out;

  char* ws = (char*)d_ws;
  size_t off = 0;
  auto alloc = [&](size_t bytes) -> void* {
    void* p = ws + off;
    off += (bytes + 255) & ~(size_t)255;
    return p;
  };
  bf16*  hq    = (bf16*)alloc((size_t)M_ * D_ * 2);   // LN1(x) bf16
  bf16*  WqT   = (bf16*)alloc((size_t)D_ * D_ * 2);   // weights bf16, [N][K]
  bf16*  WkT   = (bf16*)alloc((size_t)D_ * D_ * 2);
  bf16*  WvT   = (bf16*)alloc((size_t)D_ * D_ * 2);
  bf16*  WoT   = (bf16*)alloc((size_t)D_ * D_ * 2);
  bf16*  W1T   = (bf16*)alloc((size_t)D_ * F_ * 2);
  bf16*  W2T   = (bf16*)alloc((size_t)F_ * D_ * 2);
  bf16*  qb    = (bf16*)alloc((size_t)M_ * D_ * 2);   // pre-scaled by 1/8
  bf16*  kbuf  = (bf16*)alloc((size_t)M_ * D_ * 2);
  bf16*  vT    = (bf16*)alloc((size_t)M_ * D_ * 2);   // V transposed [1024][B*S]
  bf16*  attnb = (bf16*)alloc((size_t)M_ * D_ * 2);
  float* xres  = (float*)alloc((size_t)M_ * D_ * 4);  // x + attn@Wo
  bf16*  h2    = (bf16*)alloc((size_t)M_ * D_ * 2);   // LN2 bf16
  bf16*  ffn1  = (bf16*)alloc((size_t)M_ * F_ * 2);   // relu(h2@W1+b1)

  // one-time weight transpose-casts: f32 [K][N] -> bf16 [N][K]
  transpose_cast_bf16<<<dim3(D_ / 64, D_ / 64), 256, 0, stream>>>(Wq, WqT, D_, D_);
  transpose_cast_bf16<<<dim3(D_ / 64, D_ / 64), 256, 0, stream>>>(Wk, WkT, D_, D_);
  transpose_cast_bf16<<<dim3(D_ / 64, D_ / 64), 256, 0, stream>>>(Wv, WvT, D_, D_);
  transpose_cast_bf16<<<dim3(D_ / 64, D_ / 64), 256, 0, stream>>>(Wo, WoT, D_, D_);
  transpose_cast_bf16<<<dim3(D_ / 64, F_ / 64), 256, 0, stream>>>(W1, W1T, D_, F_);
  transpose_cast_bf16<<<dim3(F_ / 64, D_ / 64), 256, 0, stream>>>(W2, W2T, F_, D_);

  // LN1
  layernorm_bf16_k<<<M_, 256, 0, stream>>>(x, ln1s, ln1b, hq);

  // QKV projections (Q pre-scaled; V written transposed for the flash kernel)
  dim3 g1(D_ / 128, M_ / 128);
  gemm_bf16_wmma<false, false, false, true, false, true>
      <<<g1, 256, 0, stream>>>(hq, WqT, bq, nullptr, nullptr, qb,   M_, D_, D_);
  gemm_bf16_wmma<false, false, false, true, false, false>
      <<<g1, 256, 0, stream>>>(hq, WkT, bk, nullptr, nullptr, kbuf, M_, D_, D_);
  gemm_bf16_wmma<false, false, false, true, true, false>
      <<<g1, 256, 0, stream>>>(hq, WvT, bv, nullptr, nullptr, vT,   M_, D_, D_);

  // attention
  flash_attn_wmma<<<dim3(S_ / 128, H_, B_), 256, 0, stream>>>(qb, kbuf, vT, attnb);

  // Wo projection + residual -> xres (f32)
  gemm_bf16_wmma<false, true, true, false, false, false>
      <<<g1, 256, 0, stream>>>(attnb, WoT, bo, x, xres, nullptr, M_, D_, D_);

  // LN2 + FFN + residual
  layernorm_bf16_k<<<M_, 256, 0, stream>>>(xres, ln2s, ln2b, h2);
  gemm_bf16_wmma<true, false, false, true, false, false>
      <<<dim3(F_ / 128, M_ / 128), 256, 0, stream>>>(h2, W1T, b1, nullptr, nullptr, ffn1,
                                                     M_, F_, D_);
  gemm_bf16_wmma<false, true, true, false, false, false>
      <<<g1, 256, 0, stream>>>(ffn1, W2T, b2, xres, out, nullptr, M_, D_, F_);
}
